// SimpleNetHead_65798898975282
// MI455X (gfx1250) — matrix operand order repaired and checked by reading, not compiled, not run on hardware
//
#include <hip/hip_runtime.h>
#include <hip/hip_bf16.h>

// CDNA5 (gfx1250) implementation of the SASA attention head.
//
// MI455X design notes:
//  * HBM-bound workload (~17 GFLOP vs ~hundreds of MB of traffic at
//    23.3 TB/s) -> f16 intermediates, WMMA for all GEMMs.
//  * 1x1-conv GEMM: 256-thread (8-wave) workgroups. The 16-pixel x 32-channel
//    activation tile is staged in LDS with CDNA5 async-to-LDS loads
//    (global_load_async_to_lds_b32 / s_wait_asynccnt, per
//    CDNA5 ISA) and shared by all 8 waves; each wave owns
//    a 64-wide slice of Cout=512 and runs v_wmma_f32_16x16x32_f16.
//  * Weight stream prefetched with global_prefetch_b8.
//  * Attention softmax + ReLU + spatial mean: fused f32 kernel, LDS tree
//    reduction (deterministic, no float atomics).

typedef _Float16 half_t;
typedef __attribute__((ext_vector_type(16))) _Float16 v16h;
typedef __attribute__((ext_vector_type(8)))  float    v8f;

// ---------------------------------------------------------------------------
// GEMM for 1x1 conv (optionally on a zero-padded spatial grid).
//   A[m, k] = x[b, k, hi, wi]   (zero when source pixel is out of bounds)
//   B[k, n] = w[n, k]           (w is [Cout, Cin] row-major)
//   out[b, n, ho, wo] = (A x B)[m, n] + bias[n]   stored f16
// Block: 256 threads = 8 waves. Tile: 16 pixels x 512 channels
// (wave wv covers channels [wv*64, wv*64+64)). K in steps of 32.
// ---------------------------------------------------------------------------
__global__ __launch_bounds__(256)
void gemm_conv1x1_f16(const float* __restrict__ x,    // [B, Cin, H, W]
                      const float* __restrict__ w,    // [Cout, Cin]
                      const float* __restrict__ bias, // [Cout]
                      half_t* __restrict__ out,       // [B, Cout, Ho, Wo]
                      int B, int Cin, int Cout, int H, int W, int pad)
{
    const int Ho = H + 2 * pad, Wo = W + 2 * pad;
    const int HWo = Ho * Wo, HW = H * W;

    __shared__ float tileA[16 * 32];       // [pixel][k]  2 KB

    const int t    = threadIdx.x;          // 0..255
    const int lane = t & 31;
    const int wv   = t >> 5;               // wave id 0..7
    const int m0   = blockIdx.x * 16;      // 16-pixel tile
    const int n0   = wv * 64;              // this wave's channel slice

    // ---- cooperative loader role: thread t owns (px = t&15, k = t>>4, +16) -
    const int px = t & 15;
    const int kk = t >> 4;                 // 0..15
    const int mA = m0 + px;
    const int b  = mA / HWo;
    const int sp = mA % HWo;
    const int hi = sp / Wo - pad;
    const int wi = sp % Wo - pad;
    const bool inb = (hi >= 0) && (hi < H) && (wi >= 0) && (wi < W);
    const float* xb = x + ((size_t)b * Cin) * HW + (inb ? hi * W + wi : 0);

    // LDS byte offsets for the async loads (generic->LDS offset via low bits)
    const unsigned lbase = (unsigned)(uintptr_t)tileA;
    const unsigned loff0 = lbase + (unsigned)(px * 32 + kk) * 4u;
    const unsigned loff1 = loff0 + 16u * 4u;

    // Padded rows never receive async data: zero them once.
    if (!inb) {
        tileA[px * 32 + kk]      = 0.f;
        tileA[px * 32 + kk + 16] = 0.f;
    }

    // ---- WMMA fragment roles ----------------------------------------------
    const float* rowp  = &tileA[(lane & 15) * 32];  // A row m = lane&15
    const int    kbA   = (lane >> 4) * 16; // A elems 0..15 -> K=kbA+0..15
    const int    koffB = (lane >> 4) * 16; // B elems 0..15 -> K=koffB+0..15

    v8f acc[4] = {v8f{}, v8f{}, v8f{}, v8f{}};

    for (int k0 = 0; k0 < Cin; k0 += 32) {
        // -- async-stage this K-step's activation tile into LDS -------------
        if (inb) {
            const float* g0 = xb + (size_t)(k0 + kk) * HW;
            const float* g1 = xb + (size_t)(k0 + kk + 16) * HW;
            asm volatile("global_load_async_to_lds_b32 %0, %1, off"
                         :: "v"(loff0), "v"(g0) : "memory");
            asm volatile("global_load_async_to_lds_b32 %0, %1, off"
                         :: "v"(loff1), "v"(g1) : "memory");
        }
        asm volatile("s_wait_asynccnt 0x0" ::: "memory");
        __syncthreads();                   // tile visible to all 8 waves

        // -- A fragment from LDS (contiguous ds reads), convert to f16 ------
        v16h a;
#pragma unroll
        for (int j = 0; j < 8; ++j) {
            a[j]     = (half_t)rowp[kbA + j];
            a[j + 8] = (half_t)rowp[kbA + 8 + j];
        }

        // -- 4 WMMA tiles over this wave's 64 output channels ---------------
#pragma unroll
        for (int tt = 0; tt < 4; ++tt) {
            const int n = n0 + tt * 16 + (lane & 15);
            const float* wrow = w + (size_t)n * Cin + k0 + koffB;
            if (k0 + 32 < Cin) __builtin_prefetch(wrow + 32, 0, 0);
            const float4* wb = (const float4*)wrow;
            float wt[16];
            *(float4*)(wt + 0)  = wb[0];
            *(float4*)(wt + 4)  = wb[1];
            *(float4*)(wt + 8)  = wb[2];
            *(float4*)(wt + 12) = wb[3];
            v16h bf;
#pragma unroll
            for (int j = 0; j < 16; ++j) bf[j] = (half_t)wt[j];
            acc[tt] = __builtin_amdgcn_wmma_f32_16x16x32_f16(
                false, a, false, bf, (short)0, acc[tt], false, false);
        }
        __syncthreads();                   // done reading before next overwrite
    }

    // ---- store: C layout is (M = r + 8*(lane>=16), N = lane&15) ------------
#pragma unroll
    for (int tt = 0; tt < 4; ++tt) {
        const int n = n0 + tt * 16 + (lane & 15);
        const float bvv = bias[n];
#pragma unroll
        for (int r = 0; r < 8; ++r) {
            const int mm = m0 + r + 8 * (lane >> 4);
            const int bb = mm / HWo, ss = mm % HWo;
            out[((size_t)bb * Cout + n) * HWo + ss] = (half_t)(acc[tt][r] + bvv);
        }
    }
}

// ---------------------------------------------------------------------------
// Fused SASA attention + ReLU + spatial mean.
// One block per (b, c); 9-element per-channel softmax in f32, LDS tree
// reduction of the spatial sum. Deterministic (no atomics).
// ---------------------------------------------------------------------------
__global__ __launch_bounds__(256)
void attn_mean_kernel(const half_t* __restrict__ Q,    // [B, C, H, W]
                      const half_t* __restrict__ K,    // [B, C, H+2, W+2]
                      const half_t* __restrict__ V,    // [B, C, H+2, W+2]
                      const float* __restrict__ rel_h, // [C/2, 3]
                      const float* __restrict__ rel_w, // [C/2, 3]
                      float* __restrict__ feat,        // [B, C]
                      int B, int C, int H, int W, int accumulate)
{
    const int bc = blockIdx.x;
    const int c  = bc % C;
    const int Wp = W + 2;
    const size_t HWp = (size_t)(H + 2) * Wp;

    const half_t* q  = Q + (size_t)bc * H * W;
    const half_t* kk = K + (size_t)bc * HWp;
    const half_t* vv = V + (size_t)bc * HWp;

    // rel[kh*3+kw]: first channel half varies along kh, second along kw.
    float rel[9];
    const int Ch = C >> 1;
#pragma unroll
    for (int kh = 0; kh < 3; ++kh)
#pragma unroll
        for (int kw = 0; kw < 3; ++kw)
            rel[kh * 3 + kw] = (c < Ch) ? rel_h[c * 3 + kh]
                                        : rel_w[(c - Ch) * 3 + kw];

    float sum = 0.f;
    for (int s = threadIdx.x; s < H * W; s += blockDim.x) {
        const int h = s / W, wcol = s % W;
        const float qv = (float)q[s];
        float lg[9];
        float mx = -3.4e38f;
#pragma unroll
        for (int kh = 0; kh < 3; ++kh)
#pragma unroll
            for (int kw = 0; kw < 3; ++kw) {
                const int i = kh * 3 + kw;
                const float kvv = (float)kk[(h + kh) * Wp + (wcol + kw)] + rel[i];
                lg[i] = qv * kvv;
                mx = fmaxf(mx, lg[i]);
            }
        float den = 0.f, num = 0.f;
#pragma unroll
        for (int kh = 0; kh < 3; ++kh)
#pragma unroll
            for (int kw = 0; kw < 3; ++kw) {
                const int i = kh * 3 + kw;
                const float e = __expf(lg[i] - mx);
                den += e;
                num += e * (float)vv[(h + kh) * Wp + (wcol + kw)];
            }
        const float y = num / den;
        sum += (y > 0.f) ? y : 0.f;   // ReLU before the mean
    }

    __shared__ float red[256];
    red[threadIdx.x] = sum;
    __syncthreads();
    for (int st = 128; st > 0; st >>= 1) {
        if (threadIdx.x < st) red[threadIdx.x] += red[threadIdx.x + st];
        __syncthreads();
    }
    if (threadIdx.x == 0) {
        const float m = red[0] / (float)(H * W);
        feat[bc] = accumulate ? (feat[bc] + m) : m;
    }
}

// ---------------------------------------------------------------------------
// Dense classifier: out[m, n] = feat[m, :] . w[n, :] + bias[n], m < B (=4).
// M padded to 16 in the WMMA tile; one wave per 16-column N tile. Unsaved
// rows/columns depend only on their own A-row / B-column, so clamped loads
// plus a store guard are sufficient.
// ---------------------------------------------------------------------------
__global__ __launch_bounds__(32)
void dense_wmma(const float* __restrict__ feat, // [B, Kdim]
                const float* __restrict__ w,    // [N, Kdim]
                const float* __restrict__ bias, // [N]
                float* __restrict__ out,        // [B, N]
                int B, int Kdim, int N)
{
    const int lane  = threadIdx.x;
    const int n     = blockIdx.x * 16 + (lane & 15);
    const int nc    = (n < N) ? n : (N - 1);
    const int mA    = lane & 15;
    const int kbA   = (lane >> 4) * 16;
    const int koffB = (lane >> 4) * 16;

    v8f acc = {};
    for (int k0 = 0; k0 < Kdim; k0 += 32) {
        v16h a;
#pragma unroll
        for (int j = 0; j < 8; ++j) {
            const float lo = (mA < B) ? feat[(size_t)mA * Kdim + k0 + kbA + j]      : 0.f;
            const float hh = (mA < B) ? feat[(size_t)mA * Kdim + k0 + kbA + j + 8] : 0.f;
            a[j]     = (half_t)lo;
            a[j + 8] = (half_t)hh;
        }
        const float4* wb = (const float4*)(w + (size_t)nc * Kdim + k0 + koffB);
        float wt[16];
        *(float4*)(wt + 0)  = wb[0];
        *(float4*)(wt + 4)  = wb[1];
        *(float4*)(wt + 8)  = wb[2];
        *(float4*)(wt + 12) = wb[3];
        v16h bf;
#pragma unroll
        for (int j = 0; j < 16; ++j) bf[j] = (half_t)wt[j];
        acc = __builtin_amdgcn_wmma_f32_16x16x32_f16(
            false, a, false, bf, (short)0, acc, false, false);
    }

    if (n < N) {
        const float bvv = bias[n];
#pragma unroll
        for (int r = 0; r < 8; ++r) {
            const int mm = r + 8 * (lane >> 4);
            if (mm < B) out[(size_t)mm * N + n] = acc[r] + bvv;
        }
    }
}

// ---------------------------------------------------------------------------
extern "C" void kernel_launch(void* const* d_in, const int* in_sizes, int n_in,
                              void* d_out, int out_size, void* d_ws, size_t ws_size,
                              hipStream_t stream)
{
    const float* p[3] = {(const float*)d_in[0], (const float*)d_in[1], (const float*)d_in[2]};
    const float* wq = (const float*)d_in[3];
    const float* bq = (const float*)d_in[4];
    const float* wk = (const float*)d_in[5];
    const float* bk = (const float*)d_in[6];
    const float* wv = (const float*)d_in[7];
    const float* bv = (const float*)d_in[8];
    const float* rh = (const float*)d_in[9];
    const float* rw = (const float*)d_in[10];
    const float* wsp = (const float*)d_in[11];
    const float* bsp = (const float*)d_in[12];
    const float* wfm = (const float*)d_in[13];
    const float* bfm = (const float*)d_in[14];
    const float* wor = (const float*)d_in[15];
    const float* bor = (const float*)d_in[16];

    const int Bn = 4, Cin = 256, Cout = 512;

    // Scratch layout (reused across the 3 scales, processed sequentially):
    //   feat  f32 [4, 512]                      @ 0       (8 KB)
    //   Qbuf  f16 [4, 512, 64, 64]  (max size)  @ 8192
    //   Kbuf  f16 [4, 512, 66, 66]  (max size)
    //   Vbuf  f16 [4, 512, 66, 66]  (max size)
    char* wsb = (char*)d_ws;
    float*  feat = (float*)wsb;
    const size_t QMAX  = (size_t)Bn * Cout * 64 * 64;
    const size_t KVMAX = (size_t)Bn * Cout * 66 * 66;
    half_t* Qb = (half_t*)(wsb + 8192);
    half_t* Kb = Qb + QMAX;
    half_t* Vb = Kb + KVMAX;

    for (int i = 0; i < 3; ++i) {
        const int H  = 64 >> i;
        const int Mq = Bn * H * H;                 // multiple of 16 at all scales
        const int Mp = Bn * (H + 2) * (H + 2);     // multiple of 16 at all scales
        const float* wqi = wq + (size_t)i * Cout * Cin;
        const float* wki = wk + (size_t)i * Cout * Cin;
        const float* wvi = wv + (size_t)i * Cout * Cin;

        gemm_conv1x1_f16<<<Mq / 16, 256, 0, stream>>>(
            p[i], wqi, bq + i * Cout, Qb, Bn, Cin, Cout, H, H, /*pad=*/0);
        gemm_conv1x1_f16<<<Mp / 16, 256, 0, stream>>>(
            p[i], wki, bk + i * Cout, Kb, Bn, Cin, Cout, H, H, /*pad=*/1);
        gemm_conv1x1_f16<<<Mp / 16, 256, 0, stream>>>(
            p[i], wvi, bv + i * Cout, Vb, Bn, Cin, Cout, H, H, /*pad=*/1);

        attn_mean_kernel<<<Bn * Cout, 256, 0, stream>>>(
            Qb, Kb, Vb,
            rh + (size_t)i * (Cout / 2) * 3,
            rw + (size_t)i * (Cout / 2) * 3,
            feat, Bn, Cout, H, H, /*accumulate=*/i);
    }

    const int S = 64500, F = 489, O = 140;
    float* out = (float*)d_out;
    dense_wmma<<<(S + 15) / 16, 32, 0, stream>>>(feat, wsp, bsp, out, Bn, Cout, S);
    dense_wmma<<<(F + 15) / 16, 32, 0, stream>>>(feat, wfm, bfm,
                                                 out + (size_t)Bn * S, Bn, Cout, F);
    dense_wmma<<<(O + 15) / 16, 32, 0, stream>>>(feat, wor, bor,
                                                 out + (size_t)Bn * (S + F), Bn, Cout, O);
}
